// ExpValCircuitGraphModel_3_30803505447187
// MI455X (gfx1250) — matrix operand order repaired and checked by reading, not compiled, not run on hardware
//
#include <hip/hip_runtime.h>
#include <hip/hip_bf16.h>
#include <cstdint>

// ============================ WMMA helpers (gfx1250) ============================
typedef __attribute__((ext_vector_type(16))) _Float16 v16h;
typedef __attribute__((ext_vector_type(8)))  float    v8f;

__device__ __forceinline__ v8f wmma_f16(v16h a, v16h b, v8f c) {
  // D = A(16x32 f16) * B(32x16 f16) + C(16x16 f32)
  return __builtin_amdgcn_wmma_f32_16x16x32_f16(false, a, false, b, (short)0, c,
                                                false, false);
}

// ---- CDNA5 async global->LDS copy (ASYNCcnt-tracked DMA, ISA ch.8 / §15.18) ----
// Copies 16 bytes per lane: LDS[ldsAddr] = MEM[sbase + voff].
__device__ __forceinline__ void async_copy16(unsigned ldsAddr, unsigned voff,
                                             unsigned long long sbase) {
  asm volatile("global_load_async_to_lds_b128 %0, %1, %2"
               :: "v"(ldsAddr), "v"(voff), "s"(sbase) : "memory");
}
__device__ __forceinline__ void wait_async0() {
  asm volatile("s_wait_asynccnt 0" ::: "memory");
}
__device__ __forceinline__ unsigned lds_addr(const void* p) {
  return (unsigned)(uintptr_t)p;  // flat LDS address truncates to DS byte address
}

// A-fragment: 16x32 f16 tile, row-major in LDS, leading dim lda (halfs).
// Lane L<16: M=L, K 0..15 ; lanes 16-31: same M, K 16..31 (contiguous, matches B).
__device__ __forceinline__ v16h load_frag_a(const _Float16* t, int lda) {
  int lane = threadIdx.x & 31;
  int m = lane & 15, hb = (lane >> 4) * 16;
  v16h a;
#pragma unroll
  for (int r = 0; r < 8; ++r) {
    int k = hb + 2 * r;
    a[2 * r]     = t[m * lda + k];
    a[2 * r + 1] = t[m * lda + k + 1];
  }
  return a;
}

// B-fragment from row-major [K=32][N] tile in LDS (lane = column of B).
__device__ __forceinline__ v16h load_frag_b(const _Float16* t, int ldb) {
  int lane = threadIdx.x & 31;
  int n = lane & 15, kb = (lane >> 4) * 16;
  v16h b;
#pragma unroll
  for (int r = 0; r < 8; ++r) {
    b[2 * r]     = t[(kb + 2 * r) * ldb + n];
    b[2 * r + 1] = t[(kb + 2 * r + 1) * ldb + n];
  }
  return b;
}

// B-fragment when the LDS tile holds B^T rows, i.e. tile is [N][K] row-major.
__device__ __forceinline__ v16h load_frag_bt(const _Float16* t, int ldt) {
  int lane = threadIdx.x & 31;
  int n = lane & 15, kb = (lane >> 4) * 16;
  v16h b;
#pragma unroll
  for (int r = 0; r < 8; ++r) {
    b[2 * r]     = t[n * ldt + kb + 2 * r];
    b[2 * r + 1] = t[n * ldt + kb + 2 * r + 1];
  }
  return b;
}

// ============================ GEMM kernels ============================
// C[M,N] = A[M,K] * B[K,N] + bias[N].  128 threads = 4 waves; wave w owns 16 cols.
// Tiles staged via global_load_async_to_lds_b128, then converted f32->f16 in LDS.
// grid = (N/64, M/16, batch); sA/sB/sC are per-batch element strides.
__global__ void k_gemm_nn(const float* __restrict__ A, const float* __restrict__ B,
                          const float* __restrict__ bias, float* __restrict__ C,
                          int M, int N, int K,
                          long long sA, long long sB, long long sC) {
  __shared__ alignas(16) float fA[16 * 32];
  __shared__ alignas(16) float fB[32 * 64];
  __shared__ _Float16 sAt[16 * 32];
  __shared__ _Float16 sBt[32 * 64];
  const float* Ab = A + (long long)blockIdx.z * sA;
  const float* Bb = B + (long long)blockIdx.z * sB;
  float* Cb = C + (long long)blockIdx.z * sC;
  unsigned long long aBase = (unsigned long long)(uintptr_t)Ab;
  unsigned long long bBase = (unsigned long long)(uintptr_t)Bb;
  int m0 = blockIdx.y * 16, n0 = blockIdx.x * 64;
  int tid = threadIdx.x, w = tid >> 5, lane = tid & 31;
  unsigned ldsA = lds_addr(fA);
  unsigned ldsB = lds_addr(fB);
  v8f acc = {0.f, 0.f, 0.f, 0.f, 0.f, 0.f, 0.f, 0.f};
  for (int k0 = 0; k0 < K; k0 += 32) {
    __syncthreads();
    {  // A tile: 16x32 f32 = 128 chunks of 16B, one per thread
      int r = tid >> 3, c4 = (tid & 7) * 4;
      unsigned go = (unsigned)(((long long)(m0 + r) * K + (k0 + c4)) * 4);
      async_copy16(ldsA + (unsigned)tid * 16u, go, aBase);
    }
#pragma unroll
    for (int cc = 0; cc < 4; ++cc) {  // B tile: 32x64 f32 = 512 chunks
      int c = tid + cc * 128;
      int r = c >> 4, c4 = (c & 15) * 4;
      unsigned go = (unsigned)(((long long)(k0 + r) * N + (n0 + c4)) * 4);
      async_copy16(ldsB + (unsigned)c * 16u, go, bBase);
    }
    if (k0 + 32 < K)
      __builtin_prefetch(Bb + (long long)(k0 + 32 + (tid >> 4)) * N + n0 + (tid & 15) * 4, 0, 0);
    wait_async0();
    __syncthreads();
    for (int i = tid; i < 16 * 32; i += 128) sAt[i] = (_Float16)fA[i];
    for (int i = tid; i < 32 * 64; i += 128) sBt[i] = (_Float16)fB[i];
    __syncthreads();
    v16h af = load_frag_a(sAt, 32);
    v16h bf = load_frag_b(sBt + w * 16, 64);
    acc = wmma_f16(af, bf, acc);
  }
  int col = n0 + w * 16 + (lane & 15);
  int rb  = m0 + (lane >> 4) * 8;
  float bi = bias ? bias[col] : 0.f;
#pragma unroll
  for (int r = 0; r < 8; ++r)
    Cb[(long long)(rb + r) * N + col] = acc[r] + bi;
}

// C[M,N] = A[M,K] * B[N,K]^T   (both row-major). grid = (N/64, M/16, batch).
__global__ void k_gemm_nt(const float* __restrict__ A, const float* __restrict__ B,
                          float* __restrict__ C, int M, int N, int K,
                          long long sA, long long sB, long long sC) {
  __shared__ alignas(16) float fA[16 * 32];
  __shared__ alignas(16) float fB[64 * 32];
  __shared__ _Float16 sAt[16 * 32];
  __shared__ _Float16 sBt[64 * 32];
  const float* Ab = A + (long long)blockIdx.z * sA;
  const float* Bb = B + (long long)blockIdx.z * sB;
  float* Cb = C + (long long)blockIdx.z * sC;
  unsigned long long aBase = (unsigned long long)(uintptr_t)Ab;
  unsigned long long bBase = (unsigned long long)(uintptr_t)Bb;
  int m0 = blockIdx.y * 16, n0 = blockIdx.x * 64;
  int tid = threadIdx.x, w = tid >> 5, lane = tid & 31;
  unsigned ldsA = lds_addr(fA);
  unsigned ldsB = lds_addr(fB);
  v8f acc = {0.f, 0.f, 0.f, 0.f, 0.f, 0.f, 0.f, 0.f};
  for (int k0 = 0; k0 < K; k0 += 32) {
    __syncthreads();
    {  // A tile
      int r = tid >> 3, c4 = (tid & 7) * 4;
      unsigned go = (unsigned)(((long long)(m0 + r) * K + (k0 + c4)) * 4);
      async_copy16(ldsA + (unsigned)tid * 16u, go, aBase);
    }
#pragma unroll
    for (int cc = 0; cc < 4; ++cc) {  // B tile: 64 rows x 32 cols
      int c = tid + cc * 128;
      int r = c >> 3, c4 = (c & 7) * 4;
      unsigned go = (unsigned)(((long long)(n0 + r) * K + (k0 + c4)) * 4);
      async_copy16(ldsB + (unsigned)c * 16u, go, bBase);
    }
    if (k0 + 32 < K)
      __builtin_prefetch(Bb + (long long)(n0 + (tid >> 1)) * K + k0 + 32 + (tid & 1) * 16, 0, 0);
    wait_async0();
    __syncthreads();
    for (int i = tid; i < 16 * 32; i += 128) sAt[i] = (_Float16)fA[i];
    for (int i = tid; i < 64 * 32; i += 128) sBt[i] = (_Float16)fB[i];
    __syncthreads();
    v16h af = load_frag_a(sAt, 32);
    v16h bf = load_frag_bt(sBt + (w * 16) * 32, 32);
    acc = wmma_f16(af, bf, acc);
  }
  int col = n0 + w * 16 + (lane & 15);
  int rb  = m0 + (lane >> 4) * 8;
#pragma unroll
  for (int r = 0; r < 8; ++r)
    Cb[(long long)(rb + r) * N + col] = acc[r];
}

// ============================ Masked multi-head attention ============================
// One wave32 per (16-target tile, head, batch). d = 64.
// Out must be pre-filled with the skip projection; we accumulate attention into it.
// Dynamic LDS: qt(16x64) kt(16x64) vt(32x64) halfs, pb(16xn) halfs, sc(16xn) f32.
__global__ void k_attn(const float* __restrict__ Q, const float* __restrict__ K,
                       const float* __restrict__ V, float* __restrict__ Out,
                       const unsigned* __restrict__ adj,  // [B][n][words], bit t of row s = edge s->t
                       int n, int hd, int words, float scale) {
  extern __shared__ char smem[];
  _Float16* qt = (_Float16*)smem;            // 16*64
  _Float16* kt = qt + 16 * 64;               // 16*64
  _Float16* vt = kt + 16 * 64;               // 32*64
  _Float16* pb = vt + 32 * 64;               // 16*n
  float*    sc = (float*)(pb + 16 * n);      // 16*n
  int ti = blockIdx.x, h = blockIdx.y, b = blockIdx.z;
  int lane = threadIdx.x;
  long long nodeBase = (long long)b * n;
  long long qoff = (nodeBase + ti * 16) * hd + h * 64;
  for (int i = lane; i < 256; i += 32) {     // 16x64 floats as float4
    int r = i >> 4, c4 = (i & 15) * 4;
    float4 v = *(const float4*)(Q + qoff + (long long)r * hd + c4);
    _Float16* d = qt + r * 64 + c4;
    d[0] = (_Float16)v.x; d[1] = (_Float16)v.y; d[2] = (_Float16)v.z; d[3] = (_Float16)v.w;
  }
  __syncthreads();
  v16h qa0 = load_frag_a(qt, 64);
  v16h qa1 = load_frag_a(qt + 32, 64);
  int colS = lane & 15;
  int rb   = (lane >> 4) * 8;
  int tgt0 = ti * 16 + rb;               // 8-aligned -> 8 mask bits in one word
  int ntile = n >> 4;
  for (int sj = 0; sj < ntile; ++sj) {
    __syncthreads();
    long long koff = (nodeBase + sj * 16) * hd + h * 64;
    for (int i = lane; i < 256; i += 32) {
      int r = i >> 4, c4 = (i & 15) * 4;
      float4 v = *(const float4*)(K + koff + (long long)r * hd + c4);
      _Float16* d = kt + r * 64 + c4;
      d[0] = (_Float16)v.x; d[1] = (_Float16)v.y; d[2] = (_Float16)v.z; d[3] = (_Float16)v.w;
    }
    __syncthreads();
    v8f s = {0.f, 0.f, 0.f, 0.f, 0.f, 0.f, 0.f, 0.f};
    s = wmma_f16(qa0, load_frag_bt(kt, 64), s);
    s = wmma_f16(qa1, load_frag_bt(kt + 32, 64), s);
    int srcg = sj * 16 + colS;
    unsigned wbits =
        (adj[(nodeBase + srcg) * words + (tgt0 >> 5)] >> (tgt0 & 31)) & 0xffu;
#pragma unroll
    for (int r = 0; r < 8; ++r) {
      float v = ((wbits >> r) & 1u) ? s[r] * scale : -3.0e38f;
      sc[(rb + r) * n + srcg] = v;
    }
  }
  __syncthreads();
  // row softmax over sources (2 lanes per target row)
  {
    int row = lane >> 1, half = lane & 1;
    int js = half * (n >> 1), je = js + (n >> 1);
    float m = -3.4e38f;
    for (int j = js; j < je; ++j) m = fmaxf(m, sc[row * n + j]);
    m = fmaxf(m, __shfl_xor(m, 1, 32));
    float den = 0.f;
    for (int j = js; j < je; ++j) {
      float v = sc[row * n + j];
      den += (v < -1.0e37f) ? 0.f : __expf(v - m);
    }
    den += __shfl_xor(den, 1, 32);
    float inv = (den > 0.f) ? 1.f / den : 0.f;
    for (int j = js; j < je; ++j) {
      float v = sc[row * n + j];
      float e = (v < -1.0e37f) ? 0.f : __expf(v - m) * inv;
      pb[row * n + j] = (_Float16)e;
    }
  }
  __syncthreads();
  v8f o[4];
#pragma unroll
  for (int dc = 0; dc < 4; ++dc) o[dc] = (v8f){0.f,0.f,0.f,0.f,0.f,0.f,0.f,0.f};
  for (int s0 = 0; s0 < n; s0 += 32) {
    __syncthreads();
    long long voff = (nodeBase + s0) * hd + h * 64;
    for (int i = lane; i < 512; i += 32) {   // 32x64 floats as float4
      int r = i >> 4, c4 = (i & 15) * 4;
      float4 v = *(const float4*)(V + voff + (long long)r * hd + c4);
      _Float16* d = vt + r * 64 + c4;
      d[0] = (_Float16)v.x; d[1] = (_Float16)v.y; d[2] = (_Float16)v.z; d[3] = (_Float16)v.w;
    }
    __syncthreads();
    v16h pa = load_frag_a(pb + s0, n);
#pragma unroll
    for (int dc = 0; dc < 4; ++dc)
      o[dc] = wmma_f16(pa, load_frag_b(vt + dc * 16, 64), o[dc]);
  }
  long long obase = (nodeBase + ti * 16) * hd + h * 64;
#pragma unroll
  for (int dc = 0; dc < 4; ++dc) {
    int fcol = dc * 16 + colS;
#pragma unroll
    for (int r = 0; r < 8; ++r) {
      long long oi = obase + (long long)(rb + r) * hd + fcol;
      Out[oi] = Out[oi] + o[dc][r];   // skip connection already stored in Out
    }
  }
}

// ============================ Graph utility kernels ============================
__global__ void k_build_adj(const int* __restrict__ ei, int E, int n, int words,
                            unsigned* adj, unsigned* adjT) {
  int e = blockIdx.x * blockDim.x + threadIdx.x;
  if (e >= E) return;
  int src = ei[e], dst = ei[e + E];
  int b = src / n;
  int s = src - b * n;
  int t = dst - b * n;
  long long base = (long long)b * n;
  atomicOr(&adj[(base + s) * words + (t >> 5)], 1u << (t & 31));
  atomicOr(&adjT[(base + t) * words + (s >> 5)], 1u << (s & 31));
}

// masked max over in-neighbors (incl. self loop) per target node
__global__ void k_masked_max(const float* __restrict__ X, float* __restrict__ Y,
                             const unsigned* __restrict__ adjT, int n, int F, int words) {
  int t = blockIdx.x, b = blockIdx.y;
  long long nb = (long long)b * n;
  const unsigned* row = adjT + (nb + t) * words;
  for (int f = threadIdx.x; f < F; f += blockDim.x) {
    float m = -3.4e38f;
    for (int w = 0; w < words; ++w) {
      unsigned bits = row[w];
      if ((t >> 5) == w) bits |= 1u << (t & 31);
      while (bits) {
        int s = (w << 5) + __builtin_ctz(bits);
        bits &= bits - 1;
        m = fmaxf(m, X[(nb + s) * F + f]);
      }
    }
    Y[(nb + t) * F + f] = m;
  }
}

// out[row] = dot(X[row,:], w) + (biasPtr ? *biasPtr : 0)
__global__ void k_rowdot(const float* __restrict__ X, const float* __restrict__ w,
                         const float* __restrict__ biasPtr, float* __restrict__ out,
                         int rows, int F) {
  int wv = threadIdx.x >> 5, lane = threadIdx.x & 31;
  int row = blockIdx.x * (blockDim.x >> 5) + wv;
  if (row >= rows) return;
  const float* xr = X + (long long)row * F;
  float s = 0.f;
  for (int f = lane; f < F; f += 32) s += xr[f] * w[f];
#pragma unroll
  for (int off = 16; off; off >>= 1) s += __shfl_xor(s, off, 32);
  if (lane == 0) out[row] = s + (biasPtr ? biasPtr[0] : 0.f);
}

// St[b][t][s] = masked softmax over sources s of leaky_relu(ss[s]+st[t]+attb)
__global__ void k_softmax_S(const float* __restrict__ ss, const float* __restrict__ st,
                            const float* __restrict__ attb, float* __restrict__ St,
                            const unsigned* __restrict__ adjT, int n, int words) {
  __shared__ float red[128];
  int b = blockIdx.y, t = blockIdx.x, tid = threadIdx.x;
  long long nb = (long long)b * n;
  float stt = st[nb + t] + attb[0];
  const unsigned* row = adjT + (nb + t) * words;
  float m = -3.4e38f;
  for (int s = tid; s < n; s += 128) {
    unsigned bit = (row[s >> 5] >> (s & 31)) & 1u;
    if (s == t) bit = 1u;
    if (bit) {
      float r = ss[nb + s] + stt;
      r = r > 0.f ? r : 0.2f * r;
      m = fmaxf(m, r);
    }
  }
  red[tid] = m; __syncthreads();
  for (int o = 64; o; o >>= 1) { if (tid < o) red[tid] = fmaxf(red[tid], red[tid + o]); __syncthreads(); }
  m = red[0]; __syncthreads();
  float den = 0.f;
  for (int s = tid; s < n; s += 128) {
    unsigned bit = (row[s >> 5] >> (s & 31)) & 1u;
    if (s == t) bit = 1u;
    if (bit) {
      float r = ss[nb + s] + stt;
      r = r > 0.f ? r : 0.2f * r;
      den += __expf(r - m);
    }
  }
  red[tid] = den; __syncthreads();
  for (int o = 64; o; o >>= 1) { if (tid < o) red[tid] += red[tid + o]; __syncthreads(); }
  den = red[0]; __syncthreads();
  float inv = 1.f / den;
  for (int s = tid; s < n; s += 128) {
    unsigned bit = (row[s >> 5] >> (s & 31)) & 1u;
    if (s == t) bit = 1u;
    float v = 0.f;
    if (bit) {
      float r = ss[nb + s] + stt;
      r = r > 0.f ? r : 0.2f * r;
      v = __expf(r - m) * inv;
    }
    St[(nb + t) * n + s] = v;
  }
}

// LEConv fitness: sigmoid( sum_{s in N(t)} a[s] - deg[t]*b2[t] + c[t] ) (incl. self)
__global__ void k_fitness(const float* __restrict__ a, const float* __restrict__ b2v,
                          const float* __restrict__ c, float* __restrict__ fit,
                          const unsigned* __restrict__ adjT, int total, int n, int words) {
  int i = blockIdx.x * blockDim.x + threadIdx.x;
  if (i >= total) return;
  int b = i / n, t = i - b * n;
  long long nb = (long long)b * n;
  const unsigned* row = adjT + (nb + t) * words;
  float agg = 0.f, deg = 0.f;
  for (int w = 0; w < words; ++w) {
    unsigned bits = row[w];
    if ((t >> 5) == w) bits |= 1u << (t & 31);
    while (bits) {
      int s = (w << 5) + __builtin_ctz(bits);
      bits &= bits - 1;
      agg += a[nb + s];
      deg += 1.f;
    }
  }
  float z = agg - deg * b2v[i] + c[i];
  fit[i] = 1.f / (1.f + __expf(-z));
}

// descending bitonic top-k (tie -> lower index first), one block per batch
__global__ void k_topk(const float* __restrict__ fit, int n, int k,
                       int* __restrict__ idxOut, float* __restrict__ valOut) {
  extern __shared__ char sm[];
  float* key = (float*)sm;
  int*   kid = (int*)(key + n);
  int b = blockIdx.x, tid = threadIdx.x;
  for (int i = tid; i < n; i += blockDim.x) { key[i] = fit[(long long)b * n + i]; kid[i] = i; }
  __syncthreads();
  for (int ks = 2; ks <= n; ks <<= 1) {
    for (int j = ks >> 1; j > 0; j >>= 1) {
      for (int i = tid; i < n; i += blockDim.x) {
        int ixj = i ^ j;
        if (ixj > i) {
          bool desc = ((i & ks) == 0);
          float a = key[i], c = key[ixj];
          int ia = kid[i], ic = kid[ixj];
          bool aG = (a > c) || (a == c && ia < ic);
          if (desc ? !aG : aG) { key[i] = c; key[ixj] = a; kid[i] = ic; kid[ixj] = ia; }
        }
      }
      __syncthreads();
    }
  }
  for (int i = tid; i < k; i += blockDim.x) {
    idxOut[(long long)b * k + i] = kid[i];
    valOut[(long long)b * k + i] = key[i];
  }
}

// out[b,p,:] = in[b, idx[b,p], :] * (vals ? vals[b,p] : 1)
__global__ void k_gather(const float* __restrict__ in, const int* __restrict__ idx,
                         const float* __restrict__ vals, float* __restrict__ out,
                         int nIn, int k, int F) {
  int p = blockIdx.x, b = blockIdx.y;
  int src = idx[(long long)b * k + p];
  float sc = vals ? vals[(long long)b * k + p] : 1.f;
  const float* ip = in + ((long long)b * nIn + src) * F;
  float* op = out + ((long long)b * k + p) * F;
  for (int f = threadIdx.x; f < F; f += blockDim.x) op[f] = ip[f] * sc;
}

// dense A (with self loops) from bitmask
__global__ void k_expand_A(const unsigned* __restrict__ adj, float* __restrict__ Ad,
                           int n, int words, long long total) {
  long long i = (long long)blockIdx.x * blockDim.x + threadIdx.x;
  if (i >= total) return;
  long long nn = (long long)n * n;
  int b = (int)(i / nn);
  long long rem = i - (long long)b * nn;
  int r = (int)(rem / n), t = (int)(rem - (long long)r * n);
  unsigned bit = (adj[((long long)b * n + r) * words + (t >> 5)] >> (t & 31)) & 1u;
  Ad[i] = (bit || r == t) ? 1.f : 0.f;
}

// adjacency bitmasks from A2 (>0, diagonal excluded)
__global__ void k_bits_from_A(const float* __restrict__ A2, unsigned* __restrict__ adj,
                              unsigned* __restrict__ adjT, int n, int words, int total) {
  int i = blockIdx.x * blockDim.x + threadIdx.x;
  if (i >= total) return;
  int w = i % words; int rest = i / words; int r = rest % n; int b = rest / n;
  long long base = (long long)b * n;
  unsigned rowBits = 0, colBits = 0;
  for (int q = 0; q < 32; ++q) {
    int qq = (w << 5) + q;
    if (qq != r && A2[(base + r) * n + qq] > 0.f) rowBits |= 1u << q;
    if (qq != r && A2[(base + qq) * n + r] > 0.f) colBits |= 1u << q;
  }
  adj[(base + r) * words + w]  = rowBits;
  adjT[(base + r) * words + w] = colBits;
}

// mean-pool + concat + 3-layer MLP. one block (320 threads) per batch.
__global__ void k_final(const float* __restrict__ xsel, const float* __restrict__ expv,
                        const float* __restrict__ cdepth,
                        const float* __restrict__ W1, const float* __restrict__ b1,
                        const float* __restrict__ W2, const float* __restrict__ b2,
                        const float* __restrict__ W3, const float* __restrict__ b3,
                        float* __restrict__ out, int nNodes, int F, int Fin, int H, int EVn) {
  __shared__ float mg[256];
  __shared__ float h1[320];
  __shared__ float h2[320];
  int b = blockIdx.x, tid = threadIdx.x;
  if (tid < F) {
    float s = 0.f;
    for (int r = 0; r < nNodes; ++r) s += xsel[((long long)b * nNodes + r) * F + tid];
    mg[tid] = s / (float)nNodes;
  } else if (tid < F + EVn) {
    mg[tid] = expv[b * EVn + (tid - F)];
  } else if (tid == F + EVn) {
    mg[tid] = cdepth[b];
  }
  __syncthreads();
  { float s = b1[tid]; for (int i = 0; i < Fin; ++i) s += mg[i] * W1[i * H + tid]; h1[tid] = s > 0.f ? s : 0.f; }
  __syncthreads();
  { float s = b2[tid]; for (int i = 0; i < H; ++i) s += h1[i] * W2[i * H + tid]; h2[tid] = s > 0.f ? s : 0.f; }
  __syncthreads();
  if (tid < EVn) {
    float s = b3[tid];
    for (int i = 0; i < H; ++i) s += h2[i] * W3[i * EVn + tid];
    out[b * EVn + tid] = s;
  }
}

// ============================ Host orchestration ============================
extern "C" void kernel_launch(void* const* d_in, const int* in_sizes, int n_in,
                              void* d_out, int out_size, void* d_ws, size_t ws_size,
                              hipStream_t stream) {
  const int Bn = 8, n1 = 1024, n2 = 512, kk1 = 512, kk2 = 256;
  const int F0 = 32, HD1 = 320, HD2 = 192, H1 = 5, H2 = 3;
  const int w1 = n1 / 32, w2 = n2 / 32;
  const int E = Bn * n1 * 16;
  (void)in_sizes; (void)n_in; (void)out_size; (void)ws_size;

  const float* expv   = (const float*)d_in[0];
  const float* cdepth = (const float*)d_in[2];
  const float* nodes  = (const float*)d_in[3];
  const int*   ei     = (const int*)d_in[4];
  // tc1
  const float *Wq1=(const float*)d_in[6],  *bq1=(const float*)d_in[7];
  const float *Wk1=(const float*)d_in[8],  *bk1=(const float*)d_in[9];
  const float *Wv1=(const float*)d_in[10], *bv1=(const float*)d_in[11];
  const float *Ws1=(const float*)d_in[12], *bs1=(const float*)d_in[13];
  // pool1
  const float *Wlin1=(const float*)d_in[14], *blin1=(const float*)d_in[15];
  const float *attq1=(const float*)d_in[16], *attx1=(const float*)d_in[17], *attb1=(const float*)d_in[18];
  const float *leW1a=(const float*)d_in[19], *leb1a=(const float*)d_in[20];
  const float *leW2a=(const float*)d_in[21], *leW3a=(const float*)d_in[22], *leb3a=(const float*)d_in[23];
  // tc2
  const float *Wq2=(const float*)d_in[24], *bq2=(const float*)d_in[25];
  const float *Wk2=(const float*)d_in[26], *bk2=(const float*)d_in[27];
  const float *Wv2=(const float*)d_in[28], *bv2=(const float*)d_in[29];
  const float *Ws2=(const float*)d_in[30], *bs2=(const float*)d_in[31];
  // pool2
  const float *Wlin2=(const float*)d_in[32], *blin2=(const float*)d_in[33];
  const float *attq2=(const float*)d_in[34], *attx2=(const float*)d_in[35], *attb2=(const float*)d_in[36];
  const float *leW1b=(const float*)d_in[37], *leb1b=(const float*)d_in[38];
  const float *leW2b=(const float*)d_in[39], *leW3b=(const float*)d_in[40], *leb3b=(const float*)d_in[41];
  // mlp
  const float *mW1=(const float*)d_in[42], *mb1=(const float*)d_in[43];
  const float *mW2=(const float*)d_in[44], *mb2=(const float*)d_in[45];
  const float *mW3=(const float*)d_in[46], *mb3=(const float*)d_in[47];
  float* out = (float*)d_out;

  // -------- workspace bump allocator (deterministic layout, ~210 MB) --------
  char* wsb = (char*)d_ws; size_t off = 0;
  auto alloc = [&](size_t bytes) -> void* {
    void* p = wsb + off; off = (off + bytes + 255) & ~(size_t)255; return p;
  };
  unsigned* adj1  = (unsigned*)alloc((size_t)Bn * n1 * w1 * 4);
  unsigned* adjT1 = (unsigned*)alloc((size_t)Bn * n1 * w1 * 4);
  float* Q1 = (float*)alloc((size_t)Bn * n1 * HD1 * 4);
  float* K1 = (float*)alloc((size_t)Bn * n1 * HD1 * 4);
  float* V1 = (float*)alloc((size_t)Bn * n1 * HD1 * 4);
  float* X1 = (float*)alloc((size_t)Bn * n1 * HD1 * 4);
  float* xqr1 = (float*)alloc((size_t)Bn * n1 * HD1 * 4);
  float* xq1  = (float*)alloc((size_t)Bn * n1 * HD1 * 4);
  float* ss1 = (float*)alloc((size_t)Bn * n1 * 4);
  float* st1 = (float*)alloc((size_t)Bn * n1 * 4);
  float* St1 = (float*)alloc((size_t)Bn * n1 * n1 * 4);
  float* xn1 = (float*)alloc((size_t)Bn * n1 * HD1 * 4);
  float* av1 = (float*)alloc((size_t)Bn * n1 * 4);
  float* bvv1= (float*)alloc((size_t)Bn * n1 * 4);
  float* cv1 = (float*)alloc((size_t)Bn * n1 * 4);
  float* fit1= (float*)alloc((size_t)Bn * n1 * 4);
  int*   idx1= (int*)alloc((size_t)Bn * kk1 * 4);
  float* val1= (float*)alloc((size_t)Bn * kk1 * 4);
  float* xs1 = (float*)alloc((size_t)Bn * kk1 * HD1 * 4);
  float* G   = (float*)alloc((size_t)Bn * kk1 * n1 * 4);
  float* Ad  = (float*)alloc((size_t)Bn * n1 * n1 * 4);
  float* Tt  = (float*)alloc((size_t)Bn * kk1 * n1 * 4);
  float* A2  = (float*)alloc((size_t)Bn * kk1 * kk1 * 4);
  unsigned* adj2  = (unsigned*)alloc((size_t)Bn * n2 * w2 * 4);
  unsigned* adjT2 = (unsigned*)alloc((size_t)Bn * n2 * w2 * 4);
  float* Q2 = (float*)alloc((size_t)Bn * n2 * HD2 * 4);
  float* K2 = (float*)alloc((size_t)Bn * n2 * HD2 * 4);
  float* V2 = (float*)alloc((size_t)Bn * n2 * HD2 * 4);
  float* X2 = (float*)alloc((size_t)Bn * n2 * HD2 * 4);
  float* xqr2 = (float*)alloc((size_t)Bn * n2 * HD2 * 4);
  float* xq2  = (float*)alloc((size_t)Bn * n2 * HD2 * 4);
  float* ss2 = (float*)alloc((size_t)Bn * n2 * 4);
  float* st2 = (float*)alloc((size_t)Bn * n2 * 4);
  float* St2 = (float*)alloc((size_t)Bn * n2 * n2 * 4);
  float* xn2 = (float*)alloc((size_t)Bn * n2 * HD2 * 4);
  float* av2 = (float*)alloc((size_t)Bn * n2 * 4);
  float* bvv2= (float*)alloc((size_t)Bn * n2 * 4);
  float* cv2 = (float*)alloc((size_t)Bn * n2 * 4);
  float* fit2= (float*)alloc((size_t)Bn * n2 * 4);
  int*   idx2= (int*)alloc((size_t)Bn * kk2 * 4);
  float* val2= (float*)alloc((size_t)Bn * kk2 * 4);
  float* xs2 = (float*)alloc((size_t)Bn * kk2 * HD2 * 4);

  // -------- stage 0: adjacency --------
  hipMemsetAsync(adj1, 0, (size_t)Bn * n1 * w1 * 4, stream);
  hipMemsetAsync(adjT1, 0, (size_t)Bn * n1 * w1 * 4, stream);
  k_build_adj<<<(E + 255) / 256, 256, 0, stream>>>(ei, E, n1, w1, adj1, adjT1);

  // -------- TransformerConv 1 --------
  {
    dim3 g(HD1 / 64, (Bn * n1) / 16, 1);
    k_gemm_nn<<<g, 128, 0, stream>>>(nodes, Wq1, bq1, Q1, Bn * n1, HD1, F0, 0, 0, 0);
    k_gemm_nn<<<g, 128, 0, stream>>>(nodes, Wk1, bk1, K1, Bn * n1, HD1, F0, 0, 0, 0);
    k_gemm_nn<<<g, 128, 0, stream>>>(nodes, Wv1, bv1, V1, Bn * n1, HD1, F0, 0, 0, 0);
    k_gemm_nn<<<g, 128, 0, stream>>>(nodes, Ws1, bs1, X1, Bn * n1, HD1, F0, 0, 0, 0);  // skip into out buf
  }
  size_t sm1 = 8192 + 96 * (size_t)n1;  // tiles + pb(f16) + sc(f32)
  k_attn<<<dim3(n1 / 16, H1, Bn), 32, sm1, stream>>>(Q1, K1, V1, X1, adj1, n1, HD1, w1, 0.125f);

  // -------- ASAP pool 1 --------
  k_masked_max<<<dim3(n1, Bn), 128, 0, stream>>>(X1, xqr1, adjT1, n1, HD1, w1);
  k_gemm_nn<<<dim3(HD1 / 64, (Bn * n1) / 16, 1), 128, 0, stream>>>(
      xqr1, Wlin1, blin1, xq1, Bn * n1, HD1, HD1, 0, 0, 0);
  k_rowdot<<<(Bn * n1 + 3) / 4, 128, 0, stream>>>(X1, attx1, nullptr, ss1, Bn * n1, HD1);
  k_rowdot<<<(Bn * n1 + 3) / 4, 128, 0, stream>>>(xq1, attq1, nullptr, st1, Bn * n1, HD1);
  k_softmax_S<<<dim3(n1, Bn), 128, 0, stream>>>(ss1, st1, attb1, St1, adjT1, n1, w1);
  k_gemm_nn<<<dim3(HD1 / 64, n1 / 16, Bn), 128, 0, stream>>>(
      St1, X1, nullptr, xn1, n1, HD1, n1,
      (long long)n1 * n1, (long long)n1 * HD1, (long long)n1 * HD1);
  k_rowdot<<<(Bn * n1 + 3) / 4, 128, 0, stream>>>(xn1, leW1a, leb1a, av1, Bn * n1, HD1);
  k_rowdot<<<(Bn * n1 + 3) / 4, 128, 0, stream>>>(xn1, leW2a, nullptr, bvv1, Bn * n1, HD1);
  k_rowdot<<<(Bn * n1 + 3) / 4, 128, 0, stream>>>(xn1, leW3a, leb3a, cv1, Bn * n1, HD1);
  k_fitness<<<(Bn * n1 + 255) / 256, 256, 0, stream>>>(av1, bvv1, cv1, fit1, adjT1, Bn * n1, n1, w1);
  k_topk<<<Bn, n1 / 2, (size_t)n1 * 8, stream>>>(fit1, n1, kk1, idx1, val1);
  k_gather<<<dim3(kk1, Bn), 128, 0, stream>>>(xn1, idx1, val1, xs1, n1, kk1, HD1);
  k_gather<<<dim3(kk1, Bn), 256, 0, stream>>>(St1, idx1, nullptr, G, n1, kk1, n1);
  k_expand_A<<<(int)(((long long)Bn * n1 * n1 + 255) / 256), 256, 0, stream>>>(
      adj1, Ad, n1, w1, (long long)Bn * n1 * n1);
  // Tt[q,s] = sum_t G[q,t] * Ad[s,t]   (T^T = G * A^T)
  k_gemm_nt<<<dim3(n1 / 64, kk1 / 16, Bn), 128, 0, stream>>>(
      G, Ad, Tt, kk1, n1, n1,
      (long long)kk1 * n1, (long long)n1 * n1, (long long)kk1 * n1);
  // A2[p,q] = sum_s G[p,s] * Tt[q,s]   (S^T A S)
  k_gemm_nt<<<dim3(kk1 / 64, kk1 / 16, Bn), 128, 0, stream>>>(
      G, Tt, A2, kk1, kk1, n1,
      (long long)kk1 * n1, (long long)kk1 * n1, (long long)kk1 * kk1);
  k_bits_from_A<<<(Bn * n2 * w2 + 255) / 256, 256, 0, stream>>>(A2, adj2, adjT2, n2, w2, Bn * n2 * w2);

  // -------- TransformerConv 2 --------
  {
    dim3 g(HD2 / 64, (Bn * n2) / 16, 1);
    k_gemm_nn<<<g, 128, 0, stream>>>(xs1, Wq2, bq2, Q2, Bn * n2, HD2, HD1, 0, 0, 0);
    k_gemm_nn<<<g, 128, 0, stream>>>(xs1, Wk2, bk2, K2, Bn * n2, HD2, HD1, 0, 0, 0);
    k_gemm_nn<<<g, 128, 0, stream>>>(xs1, Wv2, bv2, V2, Bn * n2, HD2, HD1, 0, 0, 0);
    k_gemm_nn<<<g, 128, 0, stream>>>(xs1, Ws2, bs2, X2, Bn * n2, HD2, HD1, 0, 0, 0);
  }
  size_t sm2 = 8192 + 96 * (size_t)n2;
  k_attn<<<dim3(n2 / 16, H2, Bn), 32, sm2, stream>>>(Q2, K2, V2, X2, adj2, n2, HD2, w2, 0.125f);

  // -------- ASAP pool 2 (A2 of this pool is unused by the reference) --------
  k_masked_max<<<dim3(n2, Bn), 128, 0, stream>>>(X2, xqr2, adjT2, n2, HD2, w2);
  k_gemm_nn<<<dim3(HD2 / 64, (Bn * n2) / 16, 1), 128, 0, stream>>>(
      xqr2, Wlin2, blin2, xq2, Bn * n2, HD2, HD2, 0, 0, 0);
  k_rowdot<<<(Bn * n2 + 3) / 4, 128, 0, stream>>>(X2, attx2, nullptr, ss2, Bn * n2, HD2);
  k_rowdot<<<(Bn * n2 + 3) / 4, 128, 0, stream>>>(xq2, attq2, nullptr, st2, Bn * n2, HD2);
  k_softmax_S<<<dim3(n2, Bn), 128, 0, stream>>>(ss2, st2, attb2, St2, adjT2, n2, w2);
  k_gemm_nn<<<dim3(HD2 / 64, n2 / 16, Bn), 128, 0, stream>>>(
      St2, X2, nullptr, xn2, n2, HD2, n2,
      (long long)n2 * n2, (long long)n2 * HD2, (long long)n2 * HD2);
  k_rowdot<<<(Bn * n2 + 3) / 4, 128, 0, stream>>>(xn2, leW1b, leb1b, av2, Bn * n2, HD2);
  k_rowdot<<<(Bn * n2 + 3) / 4, 128, 0, stream>>>(xn2, leW2b, nullptr, bvv2, Bn * n2, HD2);
  k_rowdot<<<(Bn * n2 + 3) / 4, 128, 0, stream>>>(xn2, leW3b, leb3b, cv2, Bn * n2, HD2);
  k_fitness<<<(Bn * n2 + 255) / 256, 256, 0, stream>>>(av2, bvv2, cv2, fit2, adjT2, Bn * n2, n2, w2);
  k_topk<<<Bn, n2 / 2, (size_t)n2 * 8, stream>>>(fit2, n2, kk2, idx2, val2);
  k_gather<<<dim3(kk2, Bn), 128, 0, stream>>>(xn2, idx2, val2, xs2, n2, kk2, HD2);

  // -------- readout MLP --------
  k_final<<<Bn, 320, 0, stream>>>(xs2, expv, cdepth, mW1, mb1, mW2, mb2, mW3, mb3,
                                  out, kk2, HD2, HD2 + 1 + 4, 320, 4);
}
